// TPLCNet_88038239633697
// MI455X (gfx1250) — matrix operand order repaired and checked by reading, not compiled, not run on hardware
//
#include <hip/hip_runtime.h>
#include <hip/hip_bf16.h>
#include <math.h>

// ---------------- CDNA5 (gfx1250) wave32 WMMA types ----------------
typedef __attribute__((ext_vector_type(16))) __bf16 v16bf;
typedef __attribute__((ext_vector_type(8)))  __bf16 v8bf;
typedef __attribute__((ext_vector_type(8)))  float  v8f;

#define BATCH 64
#define SEQ   1024
#define CIN   128
#define HIDN  256
#define TGTN  128
#define PREDN 192
#define NSER  (BATCH * CIN)   // 8192 series for the DFT

// ---------------- workspace layout (bytes, all 256-aligned) ----------------
static constexpr size_t OFF_SB0   = 0;                                   // bf16 scales [8192, T]
static constexpr size_t SZ_SB0    = (size_t)NSER * 1024 * 2;
static constexpr size_t OFF_SB1   = OFF_SB0 + SZ_SB0;
static constexpr size_t SZ_SB1    = (size_t)NSER * 512 * 2;
static constexpr size_t OFF_SB2   = OFF_SB1 + SZ_SB1;
static constexpr size_t SZ_SB2    = (size_t)NSER * 256 * 2;
static constexpr size_t OFF_COS0  = OFF_SB2 + SZ_SB2;                    // DFT basis bf16 [Fp, T]
static constexpr size_t SZ_B0     = (size_t)1024 * 528 * 2;
static constexpr size_t OFF_SIN0  = OFF_COS0 + SZ_B0;
static constexpr size_t OFF_COS1  = OFF_SIN0 + SZ_B0;
static constexpr size_t SZ_B1     = (size_t)512 * 272 * 2;
static constexpr size_t OFF_SIN1  = OFF_COS1 + SZ_B1;
static constexpr size_t OFF_COS2  = OFF_SIN1 + SZ_B1;
static constexpr size_t SZ_B2     = (size_t)256 * 144 * 2;
static constexpr size_t OFF_SIN2  = OFF_COS2 + SZ_B2;
static constexpr size_t OFF_AMPS  = OFF_SIN2 + SZ_B2;                    // f32 amps: 528+272+144
static constexpr size_t SZ_AMPS   = 4096;
static constexpr size_t OFF_METAI = OFF_AMPS + SZ_AMPS;                  // int meta [3][9]
static constexpr size_t OFF_METAF = OFF_METAI + 128;                     // float wk [3][3]
static constexpr size_t OFF_PWB   = OFF_METAI + 256;                     // bf16 pw_w  [256,128]
static constexpr size_t OFF_WTB0  = OFF_PWB  + 65536;                    // bf16 w_t0  [192,1024]
static constexpr size_t OFF_WTB1  = OFF_WTB0 + (size_t)1024 * 192 * 2;
static constexpr size_t OFF_WTB2  = OFF_WTB1 + (size_t)512 * 192 * 2;
static constexpr size_t OFF_WCHB  = OFF_WTB2 + (size_t)256 * 192 * 2;    // bf16 w_ch [128,256]
static constexpr size_t OFF_DWO   = OFF_WCHB + 65536;                    // bf16 [B*T,128]
static constexpr size_t OFF_FUSED = OFF_DWO + (size_t)BATCH * 1024 * CIN * 2;   // f32 [16384,T]
static constexpr size_t OFF_FUSEDB= OFF_FUSED + (size_t)16384 * 1024 * 4;       // bf16 [16384,T]
static constexpr size_t OFF_PREDB = OFF_FUSEDB + (size_t)16384 * 1024 * 2;      // bf16 [12288,256]

// ------- WMMA fragments straight from global memory (wave32) -------
// A is row-major [M,K]; lane holds row (l&15), K-halves split by l>>4.
__device__ __forceinline__ v16bf frag_a(const __bf16* __restrict__ A, int lda,
                                        int m0, int k0, int l) {
  const int row = l & 15, kb = (l >> 4) << 4;
  const __bf16* p = A + (size_t)(m0 + row) * lda + k0;
  v8bf lo = *(const v8bf*)(p + kb);
  v8bf hi = *(const v8bf*)(p + kb + 8);
  __builtin_prefetch(p + 32, 0, 3);     // global_prefetch: next K-step
  return __builtin_shufflevector(lo, hi, 0, 1, 2, 3, 4, 5, 6, 7,
                                 8, 9, 10, 11, 12, 13, 14, 15);
}
// B stored K-major per output column: BT[n][k]. Lane (col = l&15) needs 16
// contiguous K values -> single 32B contiguous load, no LDS, no transpose.
__device__ __forceinline__ v16bf frag_b(const __bf16* __restrict__ BT, int ldb,
                                        int n0, int k0, int l) {
  const int col = l & 15, kb = (l >> 4) << 4;
  return *(const v16bf*)(BT + (size_t)(n0 + col) * ldb + k0 + kb);
}

#define WMMA_BF16(a, b, c) \
  __builtin_amdgcn_wmma_f32_16x16x32_bf16(false, (a), false, (b), (short)0, (c), false, false)

// ---------------- prep kernels ----------------
__global__ void k_cvt_bf16(const float* __restrict__ src, __bf16* __restrict__ dst, size_t n) {
  size_t tid = (size_t)blockIdx.x * blockDim.x + threadIdx.x;
  if (tid < n) dst[tid] = (__bf16)src[tid];
}

// DFT basis, K-major: dc[f*T + t] = cos(2*pi*f*t/T), zero-padded cols f>=Freal
__global__ void k_basis(__bf16* __restrict__ dc, __bf16* __restrict__ ds,
                        int T, int Fp, int Freal) {
  int tid = blockIdx.x * blockDim.x + threadIdx.x;
  if (tid >= T * Fp) return;
  int f = tid / T, tt = tid - f * T;
  float cv = 0.f, sv = 0.f;
  if (f < Freal) {
    int phase = (f * tt) % T;                 // exact reduction
    float ang = 6.2831853071795864f * (float)phase / (float)T;
    cv = cosf(ang); sv = sinf(ang);
  }
  dc[tid] = (__bf16)cv; ds[tid] = (__bf16)sv;
}

// x [B,T,C] -> bf16 scales [B*C, T], [B*C, T/2], [B*C, T/4]
__global__ void k_build_scales(const float* __restrict__ x, __bf16* __restrict__ sb0,
                               __bf16* __restrict__ sb1, __bf16* __restrict__ sb2) {
  int tid = blockIdx.x * blockDim.x + threadIdx.x;   // (b,u,c), c fastest
  if (tid >= BATCH * 256 * CIN) return;
  int c = tid & (CIN - 1);
  int j = tid >> 7;
  int u = j & 255;
  int b = j >> 8;
  float v[4];
#pragma unroll
  for (int q = 0; q < 4; ++q)
    v[q] = x[((size_t)b * SEQ + 4 * u + q) * CIN + c];
  size_t row = (size_t)b * CIN + c;
#pragma unroll
  for (int q = 0; q < 4; ++q) sb0[row * 1024 + 4 * u + q] = (__bf16)v[q];
  sb1[row * 512 + 2 * u + 0] = (__bf16)(0.5f * (v[0] + v[1]));
  sb1[row * 512 + 2 * u + 1] = (__bf16)(0.5f * (v[2] + v[3]));
  sb2[row * 256 + u] = (__bf16)(0.25f * (v[0] + v[1] + v[2] + v[3]));
}

// -------- DFT amplitudes: 2 M-tiles x (cos,sin) = 4 WMMAs per K-step --------
__global__ void __launch_bounds__(32)
k_dft(const __bf16* __restrict__ A, const __bf16* __restrict__ Bc,
      const __bf16* __restrict__ Bs, float* __restrict__ amps, int T, int Fp) {
  const int l = threadIdx.x;
  const int m0 = blockIdx.x * 32, n0 = blockIdx.y * 16;
  v8f re0 = {}, re1 = {}, im0 = {}, im1 = {};
  for (int k0 = 0; k0 < T; k0 += 32) {
    v16bf a0 = frag_a(A, T, m0, k0, l);
    v16bf a1 = frag_a(A, T, m0 + 16, k0, l);
    v16bf bc = frag_b(Bc, T, n0, k0, l);
    v16bf bs = frag_b(Bs, T, n0, k0, l);
    re0 = WMMA_BF16(a0, bc, re0);
    re1 = WMMA_BF16(a1, bc, re1);
    im0 = WMMA_BF16(a0, bs, im0);
    im1 = WMMA_BF16(a1, bs, im1);
  }
  float part = 0.f;
#pragma unroll
  for (int r = 0; r < 8; ++r) {
    part += sqrtf(re0[r] * re0[r] + im0[r] * im0[r]);
    part += sqrtf(re1[r] * re1[r] + im1[r] * im1[r]);
  }
  part += __shfl_xor(part, 16);
  if (l < 16) atomicAdd(&amps[n0 + l], part * (1.0f / (float)NSER));
}

// ---------------- top-k + geometry + softmax weights ----------------
__global__ void k_topk(const float* __restrict__ ampsAll, int* __restrict__ mi,
                       float* __restrict__ mf) {
  if (threadIdx.x != 0) return;
  const int scale = blockIdx.x;
  const int T = SEQ >> scale;
  const int F = (T >> 1) + 1;
  const int off = (scale == 0) ? 0 : (scale == 1 ? 528 : 800);
  const float* amps = ampsAll + off;
  int idx[3];
  for (int k = 0; k < 3; ++k) {
    float best = -1.0f; int bi = 1;
    for (int i = 0; i < F; ++i) {
      if ((k > 0 && i == idx[0]) || (k > 1 && i == idx[1])) continue;
      float v = (i == 0) ? 0.0f : amps[i];     // DC zeroed like reference
      if (v > best) { best = v; bi = i; }
    }
    idx[k] = bi;
  }
  int fq[3]; float aw[3]; float mx = -1e30f;
  for (int k = 0; k < 3; ++k) {
    fq[k] = (idx[k] < 1) ? 1 : idx[k];
    aw[k] = amps[fq[k]];
    mx = aw[k] > mx ? aw[k] : mx;
  }
  float e[3], es = 0.f;
  for (int k = 0; k < 3; ++k) { e[k] = expf(aw[k] - mx); es += e[k]; }
  for (int k = 0; k < 3; ++k) {
    int p = T / fq[k];
    int rows = (T + p - 1) / p;
    mi[scale * 9 + k * 3 + 0] = fq[k];
    mi[scale * 9 + k * 3 + 1] = p;
    mi[scale * 9 + k * 3 + 2] = rows;
    mf[scale * 3 + k] = e[k] / es;
  }
}

// ---------------- depthwise 3x3 on the period-folded grid ----------------
__global__ void k_dwconv(const __bf16* __restrict__ sb, const float* __restrict__ dww,
                         const float* __restrict__ dwb, const int* __restrict__ mi,
                         int scale, int freq, int T, __bf16* __restrict__ dwoutb) {
  size_t tid = (size_t)blockIdx.x * blockDim.x + threadIdx.x;
  if (tid >= (size_t)BATCH * T * CIN) return;
  int c = (int)(tid & (CIN - 1));
  size_t j = tid >> 7;
  int i = (int)(j % (size_t)T);
  int b = (int)(j / (size_t)T);
  int p    = mi[scale * 9 + freq * 3 + 1];
  int rows = mi[scale * 9 + freq * 3 + 2];
  int r = i / p, q = i - r * p;
  const __bf16* xr = sb + ((size_t)b * CIN + c) * T;
  float sum = dwb[c];
#pragma unroll
  for (int dr = -1; dr <= 1; ++dr) {
    int rr = r + dr;
    if (rr < 0 || rr >= rows) continue;
#pragma unroll
    for (int dq = -1; dq <= 1; ++dq) {
      int qq = q + dq;
      if (qq < 0 || qq >= p) continue;
      int ii = rr * p + qq;
      float v = (ii < T) ? (float)xr[ii] : 0.0f;   // appended pad region is zeros
      sum += dww[c * 9 + (dr + 1) * 3 + (dq + 1)] * v;
    }
  }
  dwoutb[((size_t)b * T + i) * CIN + c] = (__bf16)sum;
}

// ------- pointwise 128->256 GEMM (2x2 tiles/wave), accumulate wk -------
__global__ void __launch_bounds__(32)
k_pointwise(const __bf16* __restrict__ A, const __bf16* __restrict__ Bw,
            const float* __restrict__ pwb, const float* __restrict__ metaf,
            int scale, int freq, int T, float* __restrict__ fused) {
  const int l = threadIdx.x;
  const int m0 = blockIdx.x * 32, n0 = blockIdx.y * 32;
  v8f a00 = {}, a01 = {}, a10 = {}, a11 = {};
  for (int k0 = 0; k0 < CIN; k0 += 32) {
    v16bf fa0 = frag_a(A, CIN, m0, k0, l);
    v16bf fa1 = frag_a(A, CIN, m0 + 16, k0, l);
    v16bf fb0 = frag_b(Bw, CIN, n0, k0, l);
    v16bf fb1 = frag_b(Bw, CIN, n0 + 16, k0, l);
    a00 = WMMA_BF16(fa0, fb0, a00);
    a01 = WMMA_BF16(fa0, fb1, a01);
    a10 = WMMA_BF16(fa1, fb0, a10);
    a11 = WMMA_BF16(fa1, fb1, a11);
  }
  const float wk = metaf[scale * 3 + freq];
  const int col = l & 15, rb = (l >> 4) << 3;
  auto emit = [&](const v8f& acc, int mt, int nt) {
    int h = nt + col;
    float bias = pwb[h];
#pragma unroll
    for (int r = 0; r < 8; ++r) {
      int m = mt + rb + r;
      int b = m / T, i = m - b * T;
      fused[((size_t)(b * HIDN + h)) * T + i] += wk * (acc[r] + bias);
    }
  };
  emit(a00, m0, n0); emit(a01, m0, n0 + 16);
  emit(a10, m0 + 16, n0); emit(a11, m0 + 16, n0 + 16);
}

// ---------------- time projection: contract T -> 192 (2x2 tiles) -----------
__global__ void __launch_bounds__(32)
k_timeproj(const __bf16* __restrict__ A, const __bf16* __restrict__ Bw,
           const float* __restrict__ bt, int T, __bf16* __restrict__ predb) {
  const int l = threadIdx.x;
  const int m0 = blockIdx.x * 32, n0 = blockIdx.y * 32;
  v8f a00 = {}, a01 = {}, a10 = {}, a11 = {};
  for (int k0 = 0; k0 < T; k0 += 32) {
    v16bf fa0 = frag_a(A, T, m0, k0, l);
    v16bf fa1 = frag_a(A, T, m0 + 16, k0, l);
    v16bf fb0 = frag_b(Bw, T, n0, k0, l);
    v16bf fb1 = frag_b(Bw, T, n0 + 16, k0, l);
    a00 = WMMA_BF16(fa0, fb0, a00);
    a01 = WMMA_BF16(fa0, fb1, a01);
    a10 = WMMA_BF16(fa1, fb0, a10);
    a11 = WMMA_BF16(fa1, fb1, a11);
  }
  const int col = l & 15, rb = (l >> 4) << 3;
  auto emit = [&](const v8f& acc, int mt, int nt) {
    int pcol = nt + col;
    float bias = bt[pcol];
#pragma unroll
    for (int r = 0; r < 8; ++r) {
      int m = mt + rb + r;
      int b = m >> 8, h = m & 255;
      predb[((size_t)(b * PREDN + pcol) << 8) + h] = (__bf16)(acc[r] + bias);
    }
  };
  emit(a00, m0, n0); emit(a01, m0, n0 + 16);
  emit(a10, m0 + 16, n0); emit(a11, m0 + 16, n0 + 16);
}

// ---------------- channel projection 256->128 + scale weight ----------------
__global__ void __launch_bounds__(32)
k_chproj(const __bf16* __restrict__ A, const __bf16* __restrict__ Bw,
         const float* __restrict__ bch, const float* __restrict__ slog,
         int scale, float* __restrict__ out) {
  const int l = threadIdx.x;
  const int m0 = blockIdx.x * 32, n0 = blockIdx.y * 32;
  v8f a00 = {}, a01 = {}, a10 = {}, a11 = {};
  for (int k0 = 0; k0 < HIDN; k0 += 32) {
    v16bf fa0 = frag_a(A, HIDN, m0, k0, l);
    v16bf fa1 = frag_a(A, HIDN, m0 + 16, k0, l);
    v16bf fb0 = frag_b(Bw, HIDN, n0, k0, l);
    v16bf fb1 = frag_b(Bw, HIDN, n0 + 16, k0, l);
    a00 = WMMA_BF16(fa0, fb0, a00);
    a01 = WMMA_BF16(fa0, fb1, a01);
    a10 = WMMA_BF16(fa1, fb0, a10);
    a11 = WMMA_BF16(fa1, fb1, a11);
  }
  float s0 = slog[0], s1 = slog[1], s2 = slog[2];
  float mx = fmaxf(s0, fmaxf(s1, s2));
  float e0 = expf(s0 - mx), e1 = expf(s1 - mx), e2 = expf(s2 - mx);
  float es = e0 + e1 + e2;
  float sw = 3.0f * ((scale == 0) ? e0 : (scale == 1) ? e1 : e2) / es;
  const int col = l & 15, rb = (l >> 4) << 3;
  auto emit = [&](const v8f& acc, int mt, int nt) {
    int o = nt + col;
    float bias = bch[o];
#pragma unroll
    for (int r = 0; r < 8; ++r) {
      int m = mt + rb + r;
      out[(size_t)m * TGTN + o] += sw * (acc[r] + bias);
    }
  };
  emit(a00, m0, n0); emit(a01, m0, n0 + 16);
  emit(a10, m0 + 16, n0); emit(a11, m0 + 16, n0 + 16);
}

// ---------------- residual init: out = x[:, -192:, :] ----------------
__global__ void k_residual(const float* __restrict__ x, float* __restrict__ out) {
  int tid = blockIdx.x * blockDim.x + threadIdx.x;
  if (tid >= BATCH * PREDN * TGTN) return;
  int o = tid & (TGTN - 1);
  int j = tid >> 7;
  int p = j % PREDN;
  int b = j / PREDN;
  out[tid] = x[((size_t)b * SEQ + (SEQ - PREDN) + p) * CIN + o];
}

// ---------------- host side ----------------
extern "C" void kernel_launch(void* const* d_in, const int* in_sizes, int n_in,
                              void* d_out, int out_size, void* d_ws, size_t ws_size,
                              hipStream_t stream) {
  (void)in_sizes; (void)n_in; (void)out_size; (void)ws_size;
  const float* x    = (const float*)d_in[0];
  const float* dww  = (const float*)d_in[1];
  const float* dwb  = (const float*)d_in[2];
  const float* pww  = (const float*)d_in[3];
  const float* pwb  = (const float*)d_in[4];
  const float* wt[3] = {(const float*)d_in[5], (const float*)d_in[7], (const float*)d_in[9]};
  const float* bt[3] = {(const float*)d_in[6], (const float*)d_in[8], (const float*)d_in[10]};
  const float* wch  = (const float*)d_in[11];
  const float* bch  = (const float*)d_in[12];
  const float* slog = (const float*)d_in[13];
  float* out = (float*)d_out;

  char* ws = (char*)d_ws;
  __bf16* sb[3]   = {(__bf16*)(ws + OFF_SB0), (__bf16*)(ws + OFF_SB1), (__bf16*)(ws + OFF_SB2)};
  __bf16* bcos[3] = {(__bf16*)(ws + OFF_COS0), (__bf16*)(ws + OFF_COS1), (__bf16*)(ws + OFF_COS2)};
  __bf16* bsin[3] = {(__bf16*)(ws + OFF_SIN0), (__bf16*)(ws + OFF_SIN1), (__bf16*)(ws + OFF_SIN2)};
  float*  amps   = (float*)(ws + OFF_AMPS);
  int*    metaI  = (int*)(ws + OFF_METAI);
  float*  metaF  = (float*)(ws + OFF_METAF);
  __bf16* pwB    = (__bf16*)(ws + OFF_PWB);
  __bf16* wtB[3] = {(__bf16*)(ws + OFF_WTB0), (__bf16*)(ws + OFF_WTB1), (__bf16*)(ws + OFF_WTB2)};
  __bf16* wchB   = (__bf16*)(ws + OFF_WCHB);
  __bf16* dwob   = (__bf16*)(ws + OFF_DWO);
  float*  fused  = (float*)(ws + OFF_FUSED);
  __bf16* fusedb = (__bf16*)(ws + OFF_FUSEDB);
  __bf16* predb  = (__bf16*)(ws + OFF_PREDB);

  const int Ts[3]   = {1024, 512, 256};
  const int Fp[3]   = {528, 272, 144};
  const int Fr[3]   = {513, 257, 129};
  const int aoff[3] = {0, 528, 800};

  // bf16 weight copies (already N-major = K-contiguous per output column)
  k_cvt_bf16<<<(HIDN * CIN + 255) / 256, 256, 0, stream>>>(pww, pwB, (size_t)HIDN * CIN);
  for (int m = 0; m < 3; ++m)
    k_cvt_bf16<<<(PREDN * Ts[m] + 255) / 256, 256, 0, stream>>>(wt[m], wtB[m],
                                                                (size_t)PREDN * Ts[m]);
  k_cvt_bf16<<<(TGTN * HIDN + 255) / 256, 256, 0, stream>>>(wch, wchB, (size_t)TGTN * HIDN);
  // DFT basis tables (K-major)
  for (int m = 0; m < 3; ++m)
    k_basis<<<(Ts[m] * Fp[m] + 255) / 256, 256, 0, stream>>>(bcos[m], bsin[m], Ts[m], Fp[m], Fr[m]);
  // multi-scale bf16 inputs
  k_build_scales<<<(BATCH * 256 * CIN + 255) / 256, 256, 0, stream>>>(x, sb[0], sb[1], sb[2]);
  // amplitude spectra
  hipMemsetAsync(amps, 0, SZ_AMPS, stream);
  for (int m = 0; m < 3; ++m)
    k_dft<<<dim3(NSER / 32, Fp[m] / 16), 32, 0, stream>>>(sb[m], bcos[m], bsin[m],
                                                          amps + aoff[m], Ts[m], Fp[m]);
  k_topk<<<3, 32, 0, stream>>>(amps, metaI, metaF);
  // residual init of output (before any channel GEMM accumulates)
  k_residual<<<(BATCH * PREDN * TGTN + 255) / 256, 256, 0, stream>>>(x, out);

  for (int m = 0; m < 3; ++m) {
    const int T = Ts[m];
    hipMemsetAsync(fused, 0, (size_t)BATCH * HIDN * T * 4, stream);
    for (int k = 0; k < 3; ++k) {
      k_dwconv<<<(int)(((size_t)BATCH * T * CIN + 255) / 256), 256, 0, stream>>>(
          sb[m], dww, dwb, metaI, m, k, T, dwob);
      k_pointwise<<<dim3(BATCH * T / 32, HIDN / 32), 32, 0, stream>>>(
          dwob, pwB, pwb, metaF, m, k, T, fused);
    }
    k_cvt_bf16<<<(int)(((size_t)BATCH * HIDN * T + 255) / 256), 256, 0, stream>>>(
        fused, fusedb, (size_t)BATCH * HIDN * T);
    k_timeproj<<<dim3(BATCH * HIDN / 32, PREDN / 32), 32, 0, stream>>>(
        fusedb, wtB[m], bt[m], T, predb);
    k_chproj<<<dim3(BATCH * PREDN / 32, TGTN / 32), 32, 0, stream>>>(
        predb, wchB, bch, slog, m, out);
  }
}
